// MultiheadSelfAttention_72267119723071
// MI455X (gfx1250) — compile-verified, matrix operand order checked by reading, not run on hardware
//
#include <hip/hip_runtime.h>
#include <hip/hip_bf16.h>

// ---------------------------------------------------------------------------
// MHA (causal) for MI455X / gfx1250: bf16 WMMA 16x16x32, f32 accumulate.
// B=4, H=16, S=2048, D=1024, DK=64.  Software-pipelined (double-buffered)
// GEMM + flash attention; async global->LDS weight staging (ASYNCcnt path).
// ---------------------------------------------------------------------------

#define DMODEL 1024
#define SEQLEN 2048
#define NHEAD  16
#define DK     64
#define NBATCH 4

typedef __attribute__((ext_vector_type(16))) __bf16 v16bf;
typedef __attribute__((ext_vector_type(8)))  float  v8f;
typedef int v4i_vec __attribute__((vector_size(16)));   // matches builtin's V4i

union Frag16 {
    v16bf        v;
    uint4        q[2];
    unsigned int d[8];
};

#if __has_builtin(__builtin_amdgcn_global_load_async_to_lds_b128)
#define USE_ASYNC_LDS 1
#else
#define USE_ASYNC_LDS 0
#endif

__device__ __forceinline__ unsigned int f32_to_bf16u(float f) {
    unsigned int u = __float_as_uint(f);
    u += 0x7FFFu + ((u >> 16) & 1u);   // round-to-nearest-even
    return u >> 16;
}

__device__ __forceinline__ v8f wmma_bf16(const Frag16& a, const Frag16& b, v8f c) {
    return __builtin_amdgcn_wmma_f32_16x16x32_bf16(
        false, a.v, false, b.v, (short)0, c, false, false);
}

#if USE_ASYNC_LDS
__device__ __forceinline__ void async_b128(const unsigned short* gsrc,
                                           unsigned short* ldst) {
    __builtin_amdgcn_global_load_async_to_lds_b128(
        (__attribute__((address_space(1))) v4i_vec*)gsrc,
        (__attribute__((address_space(3))) v4i_vec*)ldst, 0, 0);
}
__device__ __forceinline__ void wait_async0() {
#if __has_builtin(__builtin_amdgcn_s_wait_asynccnt)
    __builtin_amdgcn_s_wait_asynccnt(0);
#else
    asm volatile("s_wait_asynccnt 0x0" ::: "memory");
#endif
}
#endif

// ---------------------------------------------------------------------------
// f32 -> bf16 conversion (4 elements / thread, packed 32-bit stores)
// ---------------------------------------------------------------------------
__global__ __launch_bounds__(256) void cvt_bf16_kernel(
    const float* __restrict__ src, unsigned int* __restrict__ dst, int n4)
{
    int i = blockIdx.x * blockDim.x + threadIdx.x;
    if (i >= n4) return;
    float4 f = ((const float4*)src)[i];
    unsigned int lo = f32_to_bf16u(f.x) | (f32_to_bf16u(f.y) << 16);
    unsigned int hi = f32_to_bf16u(f.z) | (f32_to_bf16u(f.w) << 16);
    ((uint2*)dst)[i] = make_uint2(lo, hi);
}

// ---------------------------------------------------------------------------
// GEMM: C[M=8192, N=1024] = A[M,1024](bf16) x W[N,1024](bf16)^T
// Wave tile 32x64 (8 WMMA / k-step), 8 waves -> 256x64 per workgroup.
// Double-buffered pipeline; stages W tiles via async->LDS when available.
// mode 0: bf16 [B,H,S,DK] (Q,K)  mode 1: bf16 [B,H,DK,S] (V^T)  mode 2: f32
// ---------------------------------------------------------------------------
__global__ __launch_bounds__(256) void gemm1024_kernel(
    const unsigned short* __restrict__ A,
    const unsigned short* __restrict__ W,
    void* __restrict__ outp, int mode)
{
    const int tid  = threadIdx.x;
    const int wave = tid >> 5;
    const int lane = tid & 31;
    const int ln16 = lane & 15;       // A row-in-tile / B,C col-in-tile
    const int h    = lane >> 4;       // lane half selects K sub-range
    const int m0   = blockIdx.x * 256 + wave * 32;
    const int n0   = blockIdx.y * 64;

    v8f acc[2][4] = {};
    const unsigned short* arow = A + (size_t)(m0 + ln16) * DMODEL + h * 8;

    Frag16 af[2][2];
    auto loadA = [&](int buf, int k0) {
        #pragma unroll
        for (int mi = 0; mi < 2; ++mi) {
            const unsigned short* ap = arow + (size_t)mi * 16 * DMODEL + k0;
            af[buf][mi].q[0] = *(const uint4*)(ap);
            af[buf][mi].q[1] = *(const uint4*)(ap + 16);
        }
    };

#if USE_ASYNC_LDS
    // ---- async global->LDS staging of the shared 64x32 W tile ----
    __shared__ __align__(16) unsigned short smB[2][64 * 32];
    const int nl = tid >> 2;             // 0..63 : n within tile
    const int kp = (tid & 3) * 8;        // 0,8,16,24 : k within tile
    const unsigned short* wstage = W + (size_t)(n0 + nl) * DMODEL + kp;

    async_b128(wstage, &smB[0][nl * 32 + kp]);
    loadA(0, 0);

    for (int step = 0; step < DMODEL / 32; ++step) {
        const int buf = step & 1;
        wait_async0();
        __syncthreads();                 // tile for this step visible to all
        if (step + 1 < DMODEL / 32) {
            async_b128(wstage + (step + 1) * 32, &smB[buf ^ 1][nl * 32 + kp]);
            loadA(buf ^ 1, (step + 1) * 32);
        }
        Frag16 bfr[4];
        #pragma unroll
        for (int t = 0; t < 4; ++t) {
            bfr[t].q[0] = *(const uint4*)&smB[buf][(t * 16 + ln16) * 32 + h * 8];
            bfr[t].q[1] = *(const uint4*)&smB[buf][(t * 16 + ln16) * 32 + 16 + h * 8];
        }
        #pragma unroll
        for (int mi = 0; mi < 2; ++mi)
            #pragma unroll
            for (int t = 0; t < 4; ++t)
                acc[mi][t] = wmma_bf16(af[buf][mi], bfr[t], acc[mi][t]);
    }
#else
    // ---- all-register double-buffered pipeline ----
    const unsigned short* wrow = W + (size_t)(n0 + ln16) * DMODEL + h * 8;
    Frag16 bfr[2][4];
    auto loadB = [&](int buf, int k0) {
        #pragma unroll
        for (int t = 0; t < 4; ++t) {
            const unsigned short* wp = wrow + (size_t)t * 16 * DMODEL + k0;
            bfr[buf][t].q[0] = *(const uint4*)(wp);
            bfr[buf][t].q[1] = *(const uint4*)(wp + 16);
        }
    };
    loadA(0, 0);
    loadB(0, 0);
    for (int k0 = 0; k0 < DMODEL; k0 += 32) {
        const int buf = (k0 >> 5) & 1;
        if (k0 + 32 < DMODEL) {
            loadA(buf ^ 1, k0 + 32);
            loadB(buf ^ 1, k0 + 32);
        }
        #pragma unroll
        for (int mi = 0; mi < 2; ++mi)
            #pragma unroll
            for (int t = 0; t < 4; ++t)
                acc[mi][t] = wmma_bf16(af[buf][mi], bfr[buf][t], acc[mi][t]);
    }
#endif

    // Store. C layout: lane (n=ln16, half h); VGPR r -> row m = r + h*8.
    #pragma unroll
    for (int mi = 0; mi < 2; ++mi) {
        #pragma unroll
        for (int t = 0; t < 4; ++t) {
            #pragma unroll
            for (int r = 0; r < 8; ++r) {
                int g = m0 + mi * 16 + h * 8 + r;   // global row (b*S + s)
                int o = n0 + t * 16 + ln16;         // output feature
                float val = acc[mi][t][r];
                if (mode == 2) {
                    ((float*)outp)[(size_t)g * DMODEL + o] = val;
                } else {
                    int b    = g >> 11;
                    int s    = g & (SEQLEN - 1);
                    int head = o >> 6;
                    int dk   = o & (DK - 1);
                    size_t idx = (mode == 0)
                        ? ((((size_t)b * NHEAD + head) * SEQLEN + s) * DK + dk)
                        : ((((size_t)b * NHEAD + head) * DK + dk) * SEQLEN + s);
                    ((unsigned short*)outp)[idx] = (unsigned short)f32_to_bf16u(val);
                }
            }
        }
    }
}

// ---------------------------------------------------------------------------
// Causal flash attention. One wave per 16-query block.
// S^T = K * Q^T  (softmax rows == lanes), O^T = V^T * P^T.
// K-fragments double-buffered one chunk ahead; V-fragments issued before
// the score WMMAs so their latency hides behind score + softmax work.
// ---------------------------------------------------------------------------
__global__ __launch_bounds__(256) void attn_kernel(
    const unsigned short* __restrict__ Q,
    const unsigned short* __restrict__ K,
    const unsigned short* __restrict__ VT,
    unsigned short* __restrict__ O)
{
    const int task = blockIdx.x * 8 + (threadIdx.x >> 5);
    const int lane = threadIdx.x & 31;
    const int qblk = task & (SEQLEN / 16 - 1);
    const int head = (task >> 7) & (NHEAD - 1);
    const int b    = task >> 11;
    const int q0   = qblk * 16;
    const int qi   = lane & 15;
    const int h    = lane >> 4;
    const int q_abs = q0 + qi;

    const unsigned short* Qb = Q  + ((size_t)b * NHEAD + head) * SEQLEN * DK;
    const unsigned short* Kb = K  + ((size_t)b * NHEAD + head) * SEQLEN * DK;
    const unsigned short* Vb = VT + ((size_t)b * NHEAD + head) * DK * SEQLEN;

    // Q^T B-fragments (lane = query column), one per 32-wide dk chunk.
    Frag16 qf[2];
    const unsigned short* qp = Qb + (size_t)(q0 + qi) * DK + h * 8;
    #pragma unroll
    for (int d0 = 0; d0 < 2; ++d0) {
        qf[d0].q[0] = *(const uint4*)(qp + d0 * 32);
        qf[d0].q[1] = *(const uint4*)(qp + d0 * 32 + 16);
    }

    Frag16 kf[2][4];                       // [buf][tk*2+d0]
    auto loadK = [&](int buf, int kc) {
        #pragma unroll
        for (int tk = 0; tk < 2; ++tk) {
            const unsigned short* kp =
                Kb + (size_t)(kc + tk * 16 + qi) * DK + h * 8;
            #pragma unroll
            for (int d0 = 0; d0 < 2; ++d0) {
                kf[buf][tk * 2 + d0].q[0] = *(const uint4*)(kp + d0 * 32);
                kf[buf][tk * 2 + d0].q[1] = *(const uint4*)(kp + d0 * 32 + 16);
            }
        }
    };

    float m_state = -3.0e38f, l_state = 0.0f;
    v8f acc[4] = {};

    const int nchunk = (q0 + 15) / 32 + 1;   // causal: chunks with kc <= q0+15
    loadK(0, 0);

    for (int c = 0; c < nchunk; ++c) {
        const int buf = c & 1;
        const int kc  = c * 32;

        // V-fragments for this chunk (used only after softmax).
        Frag16 vf[4];
        #pragma unroll
        for (int t = 0; t < 4; ++t) {
            const unsigned short* vp =
                Vb + (size_t)(t * 16 + qi) * SEQLEN + kc + h * 8;
            vf[t].q[0] = *(const uint4*)(vp);
            vf[t].q[1] = *(const uint4*)(vp + 16);
        }
        // Prefetch next chunk's K-fragments.
        if (c + 1 < nchunk) loadK(buf ^ 1, kc + 32);

        // ---- scores: two 16x16 S^T tiles (keys kc..+15, kc+16..+31) ----
        v8f s[2] = {};
        #pragma unroll
        for (int tk = 0; tk < 2; ++tk)
            #pragma unroll
            for (int d0 = 0; d0 < 2; ++d0)
                s[tk] = wmma_bf16(kf[buf][tk * 2 + d0], qf[d0], s[tk]);

        // ---- scale + causal mask + row max (row == lane's query) ----
        float rmax = -3.0e38f;
        #pragma unroll
        for (int t = 0; t < 2; ++t) {
            #pragma unroll
            for (int r = 0; r < 8; ++r) {
                int key = kc + t * 16 + h * 8 + r;
                float v = s[t][r] * 0.125f;          // 1/sqrt(64)
                v = (key > q_abs) ? -3.0e38f : v;
                s[t][r] = v;
                rmax = fmaxf(rmax, v);
            }
        }
        rmax = fmaxf(rmax, __shfl_xor(rmax, 16, 32));
        float newm = fmaxf(m_state, rmax);

        // ---- exp, row sum, pack P^T B-fragment (pure in-lane) ----
        float rsum = 0.0f;
        Frag16 pf;
        #pragma unroll
        for (int t = 0; t < 2; ++t) {
            #pragma unroll
            for (int r = 0; r < 8; r += 2) {
                float p0 = __expf(s[t][r]     - newm);
                float p1 = __expf(s[t][r + 1] - newm);
                rsum += p0 + p1;
                pf.d[t * 4 + (r >> 1)] =
                    f32_to_bf16u(p0) | (f32_to_bf16u(p1) << 16);
            }
        }
        rsum += __shfl_xor(rsum, 16, 32);

        float corr = __expf(m_state - newm);
        l_state = l_state * corr + rsum;
        m_state = newm;

        // ---- rescale accumulators (per-lane scalar) and O^T += V^T * P^T ----
        #pragma unroll
        for (int t = 0; t < 4; ++t) {
            #pragma unroll
            for (int r = 0; r < 8; ++r) acc[t][r] *= corr;
        }
        #pragma unroll
        for (int t = 0; t < 4; ++t)
            acc[t] = wmma_bf16(vf[t], pf, acc[t]);
    }

    // ---- finalize: /l, store O[b][q][head*64 + dk] (bf16) ----
    float inv = 1.0f / l_state;
    unsigned short* Ob = O + ((size_t)b * SEQLEN + q_abs) * DMODEL + head * DK;
    #pragma unroll
    for (int t = 0; t < 4; ++t) {
        #pragma unroll
        for (int r = 0; r < 8; ++r) {
            Ob[t * 16 + h * 8 + r] = (unsigned short)f32_to_bf16u(acc[t][r] * inv);
        }
    }
}

// ---------------------------------------------------------------------------
extern "C" void kernel_launch(void* const* d_in, const int* in_sizes, int n_in,
                              void* d_out, int out_size, void* d_ws, size_t ws_size,
                              hipStream_t stream) {
    const float* x  = (const float*)d_in[0];
    const float* wq = (const float*)d_in[1];
    const float* wk = (const float*)d_in[2];
    const float* wv = (const float*)d_in[3];
    const float* wo = (const float*)d_in[4];

    const size_t XE = (size_t)NBATCH * SEQLEN * DMODEL;   // 8388608
    const size_t WE = (size_t)DMODEL * DMODEL;            // 1048576

    char* ws = (char*)d_ws;
    unsigned short* xb  = (unsigned short*)ws; ws += XE * 2;
    unsigned short* wqb = (unsigned short*)ws; ws += WE * 2;
    unsigned short* wkb = (unsigned short*)ws; ws += WE * 2;
    unsigned short* wvb = (unsigned short*)ws; ws += WE * 2;
    unsigned short* wob = (unsigned short*)ws; ws += WE * 2;
    unsigned short* Qb  = (unsigned short*)ws; ws += XE * 2;
    unsigned short* Kb  = (unsigned short*)ws; ws += XE * 2;
    unsigned short* VTb = (unsigned short*)ws; ws += XE * 2;
    unsigned short* Ob  = (unsigned short*)ws; ws += XE * 2;

    // f32 -> bf16 staging
    cvt_bf16_kernel<<<(int)(XE / 4 / 256), 256, 0, stream>>>(x,  (unsigned int*)xb,  (int)(XE / 4));
    cvt_bf16_kernel<<<(int)(WE / 4 / 256), 256, 0, stream>>>(wq, (unsigned int*)wqb, (int)(WE / 4));
    cvt_bf16_kernel<<<(int)(WE / 4 / 256), 256, 0, stream>>>(wk, (unsigned int*)wkb, (int)(WE / 4));
    cvt_bf16_kernel<<<(int)(WE / 4 / 256), 256, 0, stream>>>(wv, (unsigned int*)wvb, (int)(WE / 4));
    cvt_bf16_kernel<<<(int)(WE / 4 / 256), 256, 0, stream>>>(wo, (unsigned int*)wob, (int)(WE / 4));

    // QKV projections (M = B*S = 8192, 256x64 per workgroup)
    dim3 gg(32, 16);
    gemm1024_kernel<<<gg, 256, 0, stream>>>(xb, wqb, Qb,  0);
    gemm1024_kernel<<<gg, 256, 0, stream>>>(xb, wkb, Kb,  0);
    gemm1024_kernel<<<gg, 256, 0, stream>>>(xb, wvb, VTb, 1);

    // Causal attention: B*H*(S/16) = 8192 wave-tasks / 8 waves per block
    attn_kernel<<<1024, 256, 0, stream>>>(Qb, Kb, VTb, Ob);

    // Output projection -> d_out (f32)
    gemm1024_kernel<<<gg, 256, 0, stream>>>(Ob, wob, d_out, 2);
}
